// GTOs_8005819040458
// MI455X (gfx1250) — compile-verified, matrix operand order checked by reading, not run on hardware
//
#include <hip/hip_runtime.h>
#include <hip/hip_bf16.h>

typedef _Float16     v16h __attribute__((ext_vector_type(16)));
typedef float        v8f  __attribute__((ext_vector_type(8)));
typedef unsigned int v8u  __attribute__((ext_vector_type(8)));

#define NPROBE 8000
#define NATOM  32
#define BSZ    8
#define TILES_PER_MOL   (NPROBE / 16)                 // 500 tiles of 16 probes
#define WAVES_PER_BLOCK 8                             // 256 threads, wave32
#define BLOCKS_PER_MOL  ((TILES_PER_MOL + WAVES_PER_BLOCK - 1) / WAVES_PER_BLOCK) // 63

__device__ __forceinline__ unsigned pkh(float a, float b) {
  auto p = __builtin_amdgcn_cvt_pkrtz(a, b);          // v_cvt_pk_f16_f32 -> one dword
  return __builtin_bit_cast(unsigned, p);
}

__global__ __launch_bounds__(256) void gto_wmma_kernel(
    const float* __restrict__ probes,   // [BSZ*NPROBE, 3]
    const float* __restrict__ atoms,    // [BSZ*NATOM, 3]
    const float* __restrict__ coeffs,   // [BSZ*NATOM, 23]
    const float* __restrict__ expos,    // [9]
    float* __restrict__ out)            // [BSZ*NPROBE]
{
  __shared__ float    s_atom[NATOM * 3];
  __shared__ unsigned s_coef[NATOM * 16];  // per atom: 32 features padded, f16x2 packed

  const int mol   = blockIdx.x / BLOCKS_PER_MOL;
  const int blkin = blockIdx.x % BLOCKS_PER_MOL;
  const int t     = threadIdx.x;

  // ---- stage this molecule's atoms + f16-packed (23 -> 32 padded) coeffs into LDS
  for (int i = t; i < NATOM * 3; i += 256)
    s_atom[i] = atoms[mol * NATOM * 3 + i];
  for (int d = t; d < NATOM * 16; d += 256) {
    int a = d >> 4, j = d & 15;
    int f0 = 2 * j, f1 = f0 + 1;
    float c0 = (f0 < 23) ? coeffs[(mol * NATOM + a) * 23 + f0] : 0.f;
    float c1 = (f1 < 23) ? coeffs[(mol * NATOM + a) * 23 + f1] : 0.f;
    s_coef[d] = pkh(c0, c1);
  }
  __syncthreads();

  const int wave = t >> 5;
  const int lane = t & 31;
  const int tile = blkin * WAVES_PER_BLOCK + wave;
  if (tile >= TILES_PER_MOL) return;                  // wave-uniform exit; EXEC stays full

  const bool half0 = (lane < 16);
  const int  m     = lane & 15;                       // probe row within the 16x16 tile
  const int  probe_base = mol * NPROBE + tile * 16;

  // ---- per-thread radial normalization: exp(0.5*(p*ln(2a)+ln2-lnGamma(p))), p=l+1.5
  const float LSF[9] = {0,0,0,0,1,1,1,2,2};
  const float GLN[9] = {-0.12078224f,-0.12078224f,-0.12078224f,-0.12078224f,
                         0.28468287f, 0.28468287f, 0.28468287f,
                         1.20097360f, 1.20097360f};
  float ex[9], nrm[9];
  #pragma unroll
  for (int o = 0; o < 9; ++o) {
    float e = expos[o];
    ex[o] = e;
    float power = LSF[o] + 1.5f;
    nrm[o] = __expf(0.5f * (power * __logf(2.f * e) + 0.69314718056f - GLN[o]));
  }

  const float px = probes[(probe_base + m) * 3 + 0];
  const float py = probes[(probe_base + m) * 3 + 1];
  const float pz = probes[(probe_base + m) * 3 + 2];

  const float invB  = 1.0f / 0.529177249f;
  const float LOG2E = 1.4426950408889634f;
  const float S3    = 1.7320508075688772f;

  v8f acc = {};

  for (int g = 0; g < NATOM / 2; ++g) {
    const int a0 = 2 * g, a1 = 2 * g + 1;
    const int ma = half0 ? a0 : a1;                   // each lane owns a UNIQUE pair

    // ---- pair geometry (reorder [1,2,0], Ang -> Bohr)
    float ax = s_atom[ma * 3 + 0], ay = s_atom[ma * 3 + 1], az = s_atom[ma * 3 + 2];
    float dx = (py - ay) * invB, dy = (pz - az) * invB, dz = (px - ax) * invB;
    float r2 = dx * dx + dy * dy + dz * dz;
    float r  = __builtin_amdgcn_sqrtf(r2) + 1e-8f;
    float ir = __builtin_amdgcn_rcpf(r);
    float x = dx * ir, y = dy * ir, z = dz * ir;
    float rr = r * r;

    // ---- real spherical harmonics l=0..2 ('norm' normalization)
    float sph1 = x, sph2 = y, sph3 = z;
    float sph4 = S3 * x * z, sph5 = S3 * x * y;
    float sph6 = y * y - 0.5f * (x * x + z * z);
    float sph7 = S3 * y * z, sph8 = 0.5f * S3 * (z * z - x * x);

    // ---- radials: nrm * r^l * exp(-a r^2)  (9 x v_exp_f32, the real cost)
    float rad[9];
    #pragma unroll
    for (int o = 0; o < 9; ++o) {
      float e2 = __builtin_amdgcn_exp2f(-(ex[o] * LOG2E) * rr);
      float rl = (o < 4) ? 1.f : ((o < 7) ? r : rr);
      rad[o] = nrm[o] * rl * e2;
    }

    // ---- 23 uncontracted features (RAD_IDX x SPH_IDX), padded to 24
    float f[24];
    f[0]=rad[0]; f[1]=rad[1]; f[2]=rad[2]; f[3]=rad[3];
    f[4]=rad[4]*sph1;  f[5]=rad[4]*sph2;  f[6]=rad[4]*sph3;
    f[7]=rad[5]*sph1;  f[8]=rad[5]*sph2;  f[9]=rad[5]*sph3;
    f[10]=rad[6]*sph1; f[11]=rad[6]*sph2; f[12]=rad[6]*sph3;
    f[13]=rad[7]*sph4; f[14]=rad[7]*sph5; f[15]=rad[7]*sph6;
    f[16]=rad[7]*sph7; f[17]=rad[7]*sph8;
    f[18]=rad[8]*sph4; f[19]=rad[8]*sph5; f[20]=rad[8]*sph6;
    f[21]=rad[8]*sph7; f[22]=rad[8]*sph8;
    f[23]=0.f;

    unsigned pk[12], sw[12];
    #pragma unroll
    for (int j = 0; j < 12; ++j) pk[j] = pkh(f[2 * j], f[2 * j + 1]);
    // SWAPX16: xor=0x10, or=0, and=0x1f -> exchange K-fragments between wave halves
    #pragma unroll
    for (int j = 0; j < 12; ++j)
      sw[j] = (unsigned)__builtin_amdgcn_ds_swizzle((int)pk[j], 0x401f);

    // ---- assemble A tiles (16x32 f16): lane<16 holds K{0..7,16..23}, lane>=16 K{8..15,24..31}
    v8u ua0, ua1;
    #pragma unroll
    for (int j = 0; j < 4; ++j) {
      ua0[j]     = half0 ? pk[j]     : sw[4 + j];   // atom a0: upper half borrows f[8..15]
      ua0[4 + j] = half0 ? pk[8 + j] : 0u;          // K 24..31 are padding
      ua1[j]     = half0 ? sw[j]     : pk[4 + j];   // atom a1: lower half borrows f[0..7]
      ua1[4 + j] = half0 ? sw[8 + j] : 0u;
    }

    // ---- B tiles (32x16 f16): only column 0 (lanes 0 and 16) carries coeffs
    const bool keep = (m == 0);
    const int  off  = half0 ? 0 : 4;
    v8u ub0, ub1;
    #pragma unroll
    for (int j = 0; j < 4; ++j) {
      unsigned lo0 = s_coef[a0 * 16 + off + j];
      unsigned hi0 = half0 ? s_coef[a0 * 16 + 8 + j] : 0u;
      ub0[j]     = keep ? lo0 : 0u;
      ub0[4 + j] = keep ? hi0 : 0u;
      unsigned lo1 = s_coef[a1 * 16 + off + j];
      unsigned hi1 = half0 ? s_coef[a1 * 16 + 8 + j] : 0u;
      ub1[j]     = keep ? lo1 : 0u;
      ub1[4 + j] = keep ? hi1 : 0u;
    }

    acc = __builtin_amdgcn_wmma_f32_16x16x32_f16(
        false, __builtin_bit_cast(v16h, ua0), false, __builtin_bit_cast(v16h, ub0),
        (short)0, acc, false, false);
    acc = __builtin_amdgcn_wmma_f32_16x16x32_f16(
        false, __builtin_bit_cast(v16h, ua1), false, __builtin_bit_cast(v16h, ub1),
        (short)0, acc, false, false);
  }

  // ---- D column 0: lane 0 holds probes m=0..7 in VGPR0..7, lane 16 holds m=8..15
  if (m == 0) {
    int base = probe_base + (lane >> 4) * 8;
    *(float4*)(out + base)     = make_float4(acc[0], acc[1], acc[2], acc[3]);
    *(float4*)(out + base + 4) = make_float4(acc[4], acc[5], acc[6], acc[7]);
  }
}

extern "C" void kernel_launch(void* const* d_in, const int* in_sizes, int n_in,
                              void* d_out, int out_size, void* d_ws, size_t ws_size,
                              hipStream_t stream) {
  (void)in_sizes; (void)n_in; (void)d_ws; (void)ws_size; (void)out_size;
  const float* probes = (const float*)d_in[0];  // probe_coords [64000,3]
  const float* atoms  = (const float*)d_in[1];  // atom_coords  [256,3]
  const float* coeffs = (const float*)d_in[4];  // coeffs       [256,23]
  const float* expos  = (const float*)d_in[5];  // expos        [9]
  float* out = (float*)d_out;                   // [64000] f32, every element written

  dim3 grid(BSZ * BLOCKS_PER_MOL);              // 504 blocks
  gto_wmma_kernel<<<grid, 256, 0, stream>>>(probes, atoms, coeffs, expos, out);
}
